// NodeModel_24275155157631
// MI455X (gfx1250) — matrix operand order repaired and checked by reading, not compiled, not run on hardware
//
#include <hip/hip_runtime.h>

// ---------------------------------------------------------------------------
// GraphNet NodeModel for MI455X (gfx1250): bf16 WMMA GEMM pipeline
//   + async global->LDS staging (ASYNCcnt path) + non-temporal streaming.
// ---------------------------------------------------------------------------

typedef __attribute__((ext_vector_type(16))) __bf16 v16bf;
typedef __attribute__((ext_vector_type(8)))  __bf16 v8bf;
typedef __attribute__((ext_vector_type(8)))  float  v8f;

#define SLOPE 0.01f

enum { IN_BF16 = 0, IN_EDGE = 1, IN_NODE = 2 };
enum { OUT_BF16 = 0, OUT_SCATTER = 1, OUT_F32 = 2 };

// Low 32 bits of a generic pointer to __shared__ = wave-relative LDS offset
// (ISA 10.2: LDS aperture maps addr[31:0] -> LDS_ADDR).
__device__ __forceinline__ unsigned lds_off(const void* p) {
    return (unsigned)(uintptr_t)p;
}

// ---------------------------------------------------------------------------
// Tiled GEMM: out[M, NDIM] = act(in[M, KDIM] @ W[KDIM, NDIM] + bias)
//  - weight supplied transposed+bf16: Wt[NDIM][KDIM] (B fragments from global;
//    32-48KB -> WGP$/L2 resident)
//  - INMODE: bf16 copy (async DMA to LDS), edge gather, or node concat
//  - OUTMODE: NT-store bf16, atomic f32 scatter by colIdx, or store f32
//  - STATS: per-channel sum / sum-of-squares for BatchNorm
// Block = 256 threads = 8 waves; each wave computes 16 rows x NDIM.
// ---------------------------------------------------------------------------
template<int KDIM, int NDIM, int INMODE, int OUTMODE, bool ACT, bool STATS>
__global__ __launch_bounds__(256) void gemm_tile(
    const __bf16* __restrict__ inBf,
    const float*  __restrict__ xF,
    const float*  __restrict__ eF,
    const int*    __restrict__ gidx,
    const __bf16* __restrict__ Wt,
    const float*  __restrict__ bias,
    const int*    __restrict__ colIdx,
    __bf16* __restrict__ outBf,
    float*  __restrict__ outF,
    float* __restrict__ statSum,
    float* __restrict__ statSqs,
    int Mtot)
{
    constexpr int KP = KDIM + 8;   // +16B row pad: keeps 16B align, staggers banks
    __shared__ __align__(16) __bf16 lA[128 * KP];
    __shared__ float sSum[128];
    __shared__ float sSqs[128];

    const int tid = threadIdx.x;
    const int m0  = blockIdx.x * 128;

    if (STATS) {
        for (int i = tid; i < NDIM; i += 256) { sSum[i] = 0.f; sSqs[i] = 0.f; }
    }

    // ---- stage A tile (128 rows x KDIM) into LDS as bf16 ----
    if constexpr (INMODE == IN_BF16) {
        constexpr int CH = 128 * (KDIM / 8);           // 16B chunks
        if (m0 + 128 <= Mtot) {
            // full tile: async DMA straight into LDS (ASYNCcnt-tracked)
            for (int c = tid; c < CH; c += 256) {
                int r = c / (KDIM / 8);
                int k = (c % (KDIM / 8)) * 8;
                const __bf16* g = inBf + (size_t)(m0 + r) * KDIM + k;
                unsigned l = lds_off(lA + r * KP + k);
                asm volatile("global_load_async_to_lds_b128 %0, %1, off"
                             :: "v"(l), "v"(g) : "memory");
            }
            asm volatile("s_wait_asynccnt 0x0" ::: "memory");
        } else {
            for (int c = tid; c < CH; c += 256) {
                int r  = c / (KDIM / 8);
                int k  = (c % (KDIM / 8)) * 8;
                int gm = m0 + r;
                v8bf v = {(__bf16)0.f,(__bf16)0.f,(__bf16)0.f,(__bf16)0.f,
                          (__bf16)0.f,(__bf16)0.f,(__bf16)0.f,(__bf16)0.f};
                if (gm < Mtot) v = *(const v8bf*)(inBf + (size_t)gm * KDIM + k);
                *(v8bf*)(lA + r * KP + k) = v;
            }
        }
    } else {
        constexpr int CH = 128 * (KDIM / 4);           // 4 f32 per chunk
        for (int c = tid; c < CH; c += 256) {
            int r  = c / (KDIM / 4);
            int k  = (c % (KDIM / 4)) * 4;
            int gm = m0 + r;
            float4 f = make_float4(0.f, 0.f, 0.f, 0.f);
            if (gm < Mtot) {
                if constexpr (INMODE == IN_EDGE) {
                    if (k < 64) {
                        int src = gidx[gm];
                        f = *(const float4*)(xF + (size_t)src * 64 + k);
                    } else {
                        f = *(const float4*)(eF + (size_t)gm * 64 + (k - 64));
                    }
                } else { // IN_NODE: concat x[64] || agg[128]
                    if (k < 64) f = *(const float4*)(xF + (size_t)gm * 64 + k);
                    else        f = *(const float4*)(eF + (size_t)gm * 128 + (k - 64));
                }
            }
            __bf16* p = lA + r * KP + k;
            p[0] = (__bf16)f.x; p[1] = (__bf16)f.y;
            p[2] = (__bf16)f.z; p[3] = (__bf16)f.w;
        }
    }
    __syncthreads();

    // ---- WMMA compute ----
    const int wave = tid >> 5;
    const int lane = tid & 31;
    const int lrow = lane & 15;
    const int hi   = lane >> 4;
    const int mB   = wave * 16;

    constexpr int NT = NDIM / 16;
    constexpr int KT = KDIM / 32;

    v8f acc[NT];
#pragma unroll
    for (int t = 0; t < NT; ++t) {
        v8f z = {0.f,0.f,0.f,0.f,0.f,0.f,0.f,0.f};
        acc[t] = z;
    }

#pragma unroll
    for (int kk = 0; kk < KT; ++kk) {
        // A fragment (16x32 bf16): lane = row, halves 0..7 @ k+hi*8, 8..15 @ +16
        const __bf16* pa = lA + (mB + lrow) * KP + kk * 32 + hi * 8;
        v8bf a0 = *(const v8bf*)pa;
        v8bf a1 = *(const v8bf*)(pa + 16);
        v16bf A = __builtin_shufflevector(a0, a1, 0,1,2,3,4,5,6,7,8,9,10,11,12,13,14,15);
#pragma unroll
        for (int t = 0; t < NT; ++t) {
            // B fragment (32x16 bf16): lane = column, halves 0..7 @ k+hi*8, 8..15 @ +16
            const __bf16* pb = Wt + (size_t)(t * 16 + lrow) * KDIM + kk * 32 + hi * 8;
            v8bf b0 = *(const v8bf*)pb;
            v8bf b1 = *(const v8bf*)(pb + 16);
            v16bf B = __builtin_shufflevector(b0, b1, 0,1,2,3,4,5,6,7,8,9,10,11,12,13,14,15);
            acc[t] = __builtin_amdgcn_wmma_f32_16x16x32_bf16(
                false, A, false, B, (short)0, acc[t], false, false);
        }
    }

    // ---- epilogue ----
    int scatRow[8];
    if constexpr (OUTMODE == OUT_SCATTER) {
#pragma unroll
        for (int r = 0; r < 8; ++r) {
            int gm = m0 + mB + r + hi * 8;
            scatRow[r] = (gm < Mtot) ? colIdx[gm] : -1;
        }
    }

#pragma unroll
    for (int t = 0; t < NT; ++t) {
        const int n  = t * 16 + lrow;
        const float bn = bias[n];
        float psum = 0.f, psq = 0.f;
#pragma unroll
        for (int r = 0; r < 8; ++r) {
            int gm  = m0 + mB + r + hi * 8;
            float d = acc[t][r] + bn;
            if (ACT) d = (d > 0.f) ? d : SLOPE * d;
            if (gm < Mtot) {
                if (STATS) { psum += d; psq += d * d; }
                if constexpr (OUTMODE == OUT_BF16) {
                    // streaming intermediate: non-temporal (th:NT) store
                    __bf16 hv = (__bf16)d;
                    short  sb;
                    __builtin_memcpy(&sb, &hv, 2);
                    __builtin_nontemporal_store(
                        sb, (short*)(outBf + (size_t)gm * NDIM + n));
                } else if constexpr (OUTMODE == OUT_F32) {
                    outF[(size_t)gm * NDIM + n] = d;
                } else {
                    if (scatRow[r] >= 0)
                        atomicAdd(&outF[(size_t)scatRow[r] * NDIM + n], d);
                }
            }
        }
        if (STATS) {
            atomicAdd(&sSum[n], psum);
            atomicAdd(&sSqs[n], psq);
        }
    }

    if (STATS) {
        __syncthreads();
        for (int i = tid; i < NDIM; i += 256) {
            atomicAdd(&statSum[i], sSum[i]);
            atomicAdd(&statSqs[i], sSqs[i]);
        }
    }
}

// ---------------------------------------------------------------------------
// BN stats -> affine: a = g*rsqrt(var+eps), c = beta - a*mean
// ---------------------------------------------------------------------------
__global__ void bn_affine(const float* __restrict__ sum, const float* __restrict__ sqs,
                          const float* __restrict__ g, const float* __restrict__ be,
                          float* __restrict__ a, float* __restrict__ c,
                          int n, float cnt)
{
    int i = blockIdx.x * blockDim.x + threadIdx.x;
    if (i < n) {
        float m  = sum[i] / cnt;
        float v  = sqs[i] / cnt - m * m;
        float ai = g[i] * rsqrtf(v + 1e-5f);
        a[i] = ai;
        c[i] = be[i] - ai * m;
    }
}

// ---------------------------------------------------------------------------
// Fold BN affine of the previous layer into W (and transpose+convert to bf16):
//   Wt[n][k] = a[k] * W[k][n]   (bf16)
//   bOut[n]  = b[n] + sum_k c[k] * W[k][n]
// a == nullptr -> identity (plain transpose/convert). grid = N, block = K.
// ---------------------------------------------------------------------------
__global__ void fold_weight(const float* __restrict__ W, const float* __restrict__ b,
                            const float* __restrict__ a, const float* __restrict__ c,
                            __bf16* __restrict__ Wt, float* __restrict__ bOut,
                            int K, int N)
{
    __shared__ float red[192];
    const int n = blockIdx.x;
    const int k = threadIdx.x;
    float w  = W[(size_t)k * N + n];
    float av = a ? a[k] : 1.0f;
    float cv = c ? c[k] : 0.0f;
    Wt[(size_t)n * K + k] = (__bf16)(av * w);
    red[k] = cv * w;
    __syncthreads();
    if (k == 0) {
        float s = b[n];
        for (int i = 0; i < K; ++i) s += red[i];
        bOut[n] = s;
    }
}

// ---------------------------------------------------------------------------
// Host launch
// ---------------------------------------------------------------------------
extern "C" void kernel_launch(void* const* d_in, const int* in_sizes, int n_in,
                              void* d_out, int out_size, void* d_ws, size_t ws_size,
                              hipStream_t stream)
{
    const float* x    = (const float*)d_in[0];
    const int*   ei   = (const int*)d_in[1];
    const float* ea   = (const float*)d_in[2];
    // d_in[3]=u, d_in[4]=batch : unused by this model
    const float *W1a=(const float*)d_in[5],  *b1a=(const float*)d_in[6];
    const float *g1a=(const float*)d_in[7],  *be1a=(const float*)d_in[8];
    const float *W1b=(const float*)d_in[9],  *b1b=(const float*)d_in[10];
    const float *g1b=(const float*)d_in[11], *be1b=(const float*)d_in[12];
    const float *W1c=(const float*)d_in[13], *b1c=(const float*)d_in[14];
    const float *W2a=(const float*)d_in[15], *b2a=(const float*)d_in[16];
    const float *g2a=(const float*)d_in[17], *be2a=(const float*)d_in[18];
    const float *W2b=(const float*)d_in[19], *b2b=(const float*)d_in[20];
    const float *g2b=(const float*)d_in[21], *be2b=(const float*)d_in[22];
    const float *W2c=(const float*)d_in[23], *b2c=(const float*)d_in[24];

    const int Nn = in_sizes[0] / 64;      // 100000
    const int E  = in_sizes[2] / 64;      // 1600000
    const int* row = ei;
    const int* col = ei + E;

    // ---- workspace carve-up ----
    char* ws = (char*)d_ws;
    size_t off = 0;
    auto alloc = [&](size_t bytes) -> void* {
        void* p = ws + off;
        off = (off + bytes + 255) & ~(size_t)255;
        return p;
    };
    __bf16* t1   = (__bf16*)alloc((size_t)E  * 128 * sizeof(__bf16));
    __bf16* t2   = (__bf16*)alloc((size_t)E  * 128 * sizeof(__bf16));
    float*  agg  = (float*) alloc((size_t)Nn * 128 * sizeof(float));
    __bf16* t3   = (__bf16*)alloc((size_t)Nn * 128 * sizeof(__bf16));
    __bf16* t4   = (__bf16*)alloc((size_t)Nn * 128 * sizeof(__bf16));
    __bf16* Wt1a = (__bf16*)alloc(128 * 128 * sizeof(__bf16));
    __bf16* Wt1b = (__bf16*)alloc(128 * 128 * sizeof(__bf16));
    __bf16* Wt1c = (__bf16*)alloc(128 * 128 * sizeof(__bf16));
    __bf16* Wt2a = (__bf16*)alloc(192 * 128 * sizeof(__bf16));
    __bf16* Wt2b = (__bf16*)alloc(128 * 128 * sizeof(__bf16));
    __bf16* Wt2c = (__bf16*)alloc(128 * 64  * sizeof(__bf16));
    float*  bf   = (float*) alloc(6 * 128 * sizeof(float));   // folded biases
    float*  st   = (float*) alloc(8 * 128 * sizeof(float));   // 4 levels x (sum,sqs)
    float*  af   = (float*) alloc(8 * 128 * sizeof(float));   // 4 levels x (a,c)
    if (off > ws_size) return; // workspace too small; nothing sane to do

    float *b1aF = bf + 0*128, *b1bF = bf + 1*128, *b1cF = bf + 2*128;
    float *b2aF = bf + 3*128, *b2bF = bf + 4*128, *b2cF = bf + 5*128;
    float *s1 = st + 0*256, *q1 = st + 0*256 + 128;
    float *s2 = st + 1*256, *q2 = st + 1*256 + 128;
    float *s3 = st + 2*256, *q3 = st + 2*256 + 128;
    float *s4 = st + 3*256, *q4 = st + 3*256 + 128;
    float *a1 = af + 0*256, *c1 = af + 0*256 + 128;
    float *a2 = af + 1*256, *c2 = af + 1*256 + 128;
    float *a3 = af + 2*256, *c3 = af + 2*256 + 128;
    float *a4 = af + 3*256, *c4 = af + 3*256 + 128;

    hipMemsetAsync(agg, 0, (size_t)Nn * 128 * sizeof(float), stream);
    hipMemsetAsync(st,  0, 8 * 128 * sizeof(float), stream);

    const dim3 blk(256);
    const int gE = (E  + 127) / 128;
    const int gN = (Nn + 127) / 128;

    // first-layer weights: plain transpose+bf16 convert (identity fold)
    fold_weight<<<128, 128, 0, stream>>>(W1a, b1a, nullptr, nullptr, Wt1a, b1aF, 128, 128);
    fold_weight<<<128, 192, 0, stream>>>(W2a, b2a, nullptr, nullptr, Wt2a, b2aF, 192, 128);

    // ---- edge MLP ----
    gemm_tile<128,128,IN_EDGE,OUT_BF16,true,true><<<gE, blk, 0, stream>>>(
        nullptr, x, ea, row, Wt1a, b1aF, nullptr, t1, nullptr, s1, q1, E);
    bn_affine<<<1, 128, 0, stream>>>(s1, q1, g1a, be1a, a1, c1, 128, (float)E);
    fold_weight<<<128, 128, 0, stream>>>(W1b, b1b, a1, c1, Wt1b, b1bF, 128, 128);

    gemm_tile<128,128,IN_BF16,OUT_BF16,true,true><<<gE, blk, 0, stream>>>(
        t1, nullptr, nullptr, nullptr, Wt1b, b1bF, nullptr, t2, nullptr, s2, q2, E);
    bn_affine<<<1, 128, 0, stream>>>(s2, q2, g1b, be1b, a2, c2, 128, (float)E);
    fold_weight<<<128, 128, 0, stream>>>(W1c, b1c, a2, c2, Wt1c, b1cF, 128, 128);

    // final edge layer fused with scatter-add into agg (f32 atomics, L2-resident)
    gemm_tile<128,128,IN_BF16,OUT_SCATTER,false,false><<<gE, blk, 0, stream>>>(
        t2, nullptr, nullptr, nullptr, Wt1c, b1cF, col, nullptr, agg, nullptr, nullptr, E);

    // ---- node MLP ----
    gemm_tile<192,128,IN_NODE,OUT_BF16,true,true><<<gN, blk, 0, stream>>>(
        nullptr, x, agg, nullptr, Wt2a, b2aF, nullptr, t3, nullptr, s3, q3, Nn);
    bn_affine<<<1, 128, 0, stream>>>(s3, q3, g2a, be2a, a3, c3, 128, (float)Nn);
    fold_weight<<<128, 128, 0, stream>>>(W2b, b2b, a3, c3, Wt2b, b2bF, 128, 128);

    gemm_tile<128,128,IN_BF16,OUT_BF16,true,true><<<gN, blk, 0, stream>>>(
        t3, nullptr, nullptr, nullptr, Wt2b, b2bF, nullptr, t4, nullptr, s4, q4, Nn);
    bn_affine<<<1, 128, 0, stream>>>(s4, q4, g2b, be2b, a4, c4, 128, (float)Nn);
    fold_weight<<<64, 128, 0, stream>>>(W2c, b2c, a4, c4, Wt2c, b2cF, 128, 64);

    gemm_tile<128,64,IN_BF16,OUT_F32,false,false><<<gN, blk, 0, stream>>>(
        t4, nullptr, nullptr, nullptr, Wt2c, b2cF, nullptr, nullptr, (float*)d_out,
        nullptr, nullptr, Nn);
}
